// Model_21809843929350
// MI455X (gfx1250) — compile-verified, matrix operand order checked by reading, not run on hardware
//
#include <hip/hip_runtime.h>
#include <hip/hip_bf16.h>

// Problem constants
#define BATCH   1024
#define SEQ_T   512
#define FEAT    38
#define HID     128
#define GATES   512      // 4*HID
#define KCAT    192      // logical K: 128 (h) + 64 (x padded 38->64)
#define KCATP   200      // padded row stride (bf16) for W and aBuf: 100 dwords -> conflict-free b128
#define KFC     128
#define KFCP    136      // padded Wfc row stride (bf16): 68 dwords -> conflict-free
#define NFC     48       // fc output padded 38->48 (3 WMMA tiles)
#define GSTR    20       // gates row stride (floats): 20 dwords -> conflict-free b128
#define MTILE   16       // batch rows per workgroup
#define TOUT    (SEQ_T - 1)

typedef __attribute__((ext_vector_type(16))) __bf16   v16bf;
typedef __attribute__((ext_vector_type(2)))  __bf16   v2bf;
typedef __attribute__((ext_vector_type(8)))  float    v8f;
typedef __attribute__((ext_vector_type(4)))  float    v4f;
typedef __attribute__((ext_vector_type(4)))  unsigned u32x4;

union FragU { u32x4 q[2]; v16bf v; };
union AccU  { v8f a; v4f h[2]; };

// Native tanh on CDNA5 (V_TANH_F32 is a hardware TRANS op); guarded fallbacks.
#if __has_builtin(__builtin_amdgcn_tanhf)
static __device__ __forceinline__ float fast_tanh(float x) { return __builtin_amdgcn_tanhf(x); }
#elif __has_builtin(__builtin_amdgcn_tanh_f32)
static __device__ __forceinline__ float fast_tanh(float x) { return __builtin_amdgcn_tanh_f32(x); }
#else
static __device__ __forceinline__ float fast_tanh(float x) { return tanhf(x); }
#endif
// sigmoid(x) = 0.5*tanh(x/2) + 0.5  -> one TRANS + two FMAs
static __device__ __forceinline__ float fast_sig(float x) {
    return __builtin_fmaf(0.5f, fast_tanh(0.5f * x), 0.5f);
}

// Load a 16x32 bf16 WMMA operand fragment row-slice from LDS.
// CDNA5 16-bit A/B layout: lanes 0-15 hold K = kOff+0..7 and kOff+16..23;
// lanes 16-31 hold the +8-shifted halves -> two contiguous 16B chunks per lane.
static __device__ __forceinline__ v16bf ldsFrag(const __bf16* p) {
    FragU u;
    u.q[0] = *(const u32x4*)(p);
    u.q[1] = *(const u32x4*)(p + 16);
    return u.v;
}

struct alignas(16) SharedMem {
    __bf16 W[GATES * KCATP];     // 204800 B  packed [W_hh | W_ih | 0] bf16, row n, padded stride
    __bf16 Wfc[NFC * KFCP];      //  13056 B  fc weights padded to 48 rows
    __bf16 aBuf[MTILE * KCATP];  //   6400 B  A operand rows: [h bf16 | x bf16 | 0]
    float  gates[GATES * GSTR];  //  40960 B  pre-activations, TRANSPOSED [n][m], padded stride
    float  cst[HID * MTILE];     //   8192 B  cell state, [j][r] layout
    float  bias[GATES];          //   2048 B
    float  bfc[NFC];             //    192 B
};                               // total ~275.6 KB < 320 KB/WGP

__global__ __launch_bounds__(128, 1)
void lstm_seq2seq_kernel(const float* __restrict__ inputs,
                         const float* __restrict__ W_ih_e, const float* __restrict__ W_hh_e,
                         const float* __restrict__ b_e,
                         const float* __restrict__ W_ih_d, const float* __restrict__ W_hh_d,
                         const float* __restrict__ b_d,
                         const float* __restrict__ W_fc,   const float* __restrict__ b_fc,
                         float* __restrict__ out)
{
    __shared__ SharedMem s;
    const int tid   = threadIdx.x;
    const int wave  = tid >> 5;       // 4 waves: each owns one gate (128 gate columns)
    const int lane  = tid & 31;
    const int mrow  = lane & 15;
    const int khalf = lane >> 4;
    const int r0    = blockIdx.x * MTILE;   // this WG's batch rows

    // ---- pack [W_hh | W_ih | 0] -> bf16 LDS (padded rows), and bias ----
    auto packWeights = [&](const float* Wih, const float* Whh, const float* b) {
        for (int i = tid; i < GATES * KCAT; i += 128) {
            int n = i / KCAT, k = i - n * KCAT;
            float w = 0.0f;
            if (k < HID)               w = Whh[n * HID + k];
            else if ((k - HID) < FEAT) w = Wih[n * FEAT + (k - HID)];
            s.W[n * KCATP + k] = (__bf16)w;
        }
        for (int i = tid; i < GATES; i += 128) s.bias[i] = b[i];
    };

    // ---- prefetch x_t (16x64, zero-padded) into registers; commit later ----
    auto prefetchX = [&](int t, float* xr) {
        #pragma unroll
        for (int u = 0; u < 8; ++u) {
            int i = tid + 128 * u;
            int r = i >> 6, f = i & 63;
            xr[u] = (f < FEAT)
                ? inputs[(size_t)(r0 + r) * (SEQ_T * FEAT) + (size_t)t * FEAT + f] : 0.0f;
        }
    };
    auto commitX = [&](const float* xr) {
        #pragma unroll
        for (int u = 0; u < 8; ++u) {
            int i = tid + 128 * u;
            int r = i >> 6, f = i & 63;
            s.aBuf[r * KCATP + HID + f] = (__bf16)xr[u];
        }
    };

    // ---- gates = aBuf(16x192) @ W^T via bf16 WMMA; wave w -> gate cols [128w,128w+128) ----
    // B fragments software-pipelined 2 deep so WMMA waits leave 4 chunk-loads in flight.
    auto gateMatmul = [&]() {
        v16bf afrag[6];
        #pragma unroll
        for (int kb = 0; kb < 6; ++kb)
            afrag[kb] = ldsFrag(&s.aBuf[mrow * KCATP] + kb * 32 + khalf * 8);

        const __bf16* wbase = &s.W[(wave * 128 + mrow) * KCATP + khalf * 8];
        // fragment idx = nt*6 + kb -> LDS address
        auto baddr = [&](int idx) {
            int nt = idx / 6, kb = idx - nt * 6;
            return wbase + (nt * 16) * KCATP + kb * 32;
        };

        v16bf b0 = ldsFrag(baddr(0));
        v16bf b1 = ldsFrag(baddr(1));
        v8f acc = {};
        #pragma unroll
        for (int idx = 0; idx < 48; ++idx) {
            const int nt = idx / 6, kb = idx - nt * 6;
            v16bf bcur = (idx & 1) ? b1 : b0;
            if (idx + 2 < 48) {               // prefetch 2 fragments ahead
                v16bf bn = ldsFrag(baddr(idx + 2));
                if (idx & 1) b1 = bn; else b0 = bn;
            }
            acc = __builtin_amdgcn_wmma_f32_16x16x32_bf16(
                      false, afrag[kb], false, bcur, (short)0, acc, false, false);
            if (kb == 5) {
                // D layout: VGPR v -> row (v + 8*khalf), col (n0 + lane%16).
                // Transposed gate buffer: 8 values contiguous -> two ds_store_b128.
                AccU u; u.a = acc;
                float* gp = &s.gates[(wave * 128 + nt * 16 + mrow) * GSTR + 8 * khalf];
                *(v4f*)(gp)     = u.h[0];
                *(v4f*)(gp + 4) = u.h[1];
                acc = (v8f){};
            }
        }
    };

    // ---- LSTM pointwise: c = sig(f)*c + sig(i)*tanh(g); h = sig(o)*tanh(c) ----
    auto gateElem = [&]() {
        for (int i = tid; i < MTILE * HID / 2; i += 128) {
            int r = i & 15;
            int j = (i >> 4) << 1;          // even column pair j, j+1
            float hpk[2];
            #pragma unroll
            for (int p = 0; p < 2; ++p) {
                int jj = j + p;
                float gi = s.gates[(      jj) * GSTR + r] + s.bias[      jj];
                float gf = s.gates[(128 + jj) * GSTR + r] + s.bias[128 + jj];
                float gg = s.gates[(256 + jj) * GSTR + r] + s.bias[256 + jj];
                float go = s.gates[(384 + jj) * GSTR + r] + s.bias[384 + jj];
                float c  = fast_sig(gf) * s.cst[jj * 16 + r] + fast_sig(gi) * fast_tanh(gg);
                hpk[p]   = fast_sig(go) * fast_tanh(c);
                s.cst[jj * 16 + r] = c;
            }
            v2bf hp = { (__bf16)hpk[0], (__bf16)hpk[1] };
            *(v2bf*)(&s.aBuf[r * KCATP + j]) = hp;   // h feeds next step's A operand
        }
    };

    // ---- init: h = c = 0, pads zero, fc weights resident for whole kernel ----
    for (int i = tid; i < MTILE * KCATP; i += 128) s.aBuf[i] = (__bf16)0.0f;
    for (int i = tid; i < MTILE * HID;   i += 128) s.cst[i]  = 0.0f;
    for (int i = tid; i < NFC * KFC; i += 128) {
        int f = i / KFC, k = i - f * KFC;
        s.Wfc[f * KFCP + k] = (f < FEAT) ? (__bf16)W_fc[f * KFC + k] : (__bf16)0.0f;
    }
    for (int i = tid; i < NFC; i += 128) s.bfc[i] = (i < FEAT) ? b_fc[i] : 0.0f;
    packWeights(W_ih_e, W_hh_e, b_e);
    {   // stage x_0
        float xr[8];
        prefetchX(0, xr);
        commitX(xr);
    }
    __syncthreads();

    // ================= encoder: 512 dependent steps, 2 barriers/step =================
    for (int t = 0; t < SEQ_T; ++t) {
        float xr[8];
        prefetchX((t + 1) & (SEQ_T - 1), xr);   // t=511 stages x_0 = decoder out0
        gateMatmul();
        __syncthreads();
        gateElem();
        commitX(xr);
        __syncthreads();
    }

    // phase switch: decoder weights overwrite encoder weights (aBuf x slot = out0 already)
    packWeights(W_ih_d, W_hh_d, b_d);
    __syncthreads();

    // ================= decoder: 511 dependent steps =================
    for (int t = 0; t < TOUT; ++t) {
        gateMatmul();
        __syncthreads();
        gateElem();
        __syncthreads();
        // y(16x38) = h(16x128) @ W_fc^T + b_fc  — waves 0..2, one 16-col tile each
        if (wave < 3) {
            const __bf16* wrow = &s.Wfc[(wave * 16 + mrow) * KFCP + khalf * 8];
            const __bf16* arow = &s.aBuf[mrow * KCATP + khalf * 8];
            v8f acc = {};
            #pragma unroll
            for (int kb = 0; kb < 4; ++kb) {
                v16bf af = ldsFrag(arow + kb * 32);
                v16bf bf = ldsFrag(wrow + kb * 32);
                acc = __builtin_amdgcn_wmma_f32_16x16x32_bf16(
                          false, af, false, bf, (short)0, acc, false, false);
            }
            const int ncol = wave * 16 + mrow;
            if (ncol < FEAT) {
                float bb = s.bfc[ncol];
                #pragma unroll
                for (int v = 0; v < 8; ++v) {
                    int m = v + 8 * khalf;
                    float y = acc[v] + bb;
                    out[(size_t)(r0 + m) * (TOUT * FEAT) + (size_t)t * FEAT + ncol] = y;
                    s.aBuf[m * KCATP + HID + ncol] = (__bf16)y;  // feedback input
                }
            }
        }
        __syncthreads();
    }
}

extern "C" void kernel_launch(void* const* d_in, const int* in_sizes, int n_in,
                              void* d_out, int out_size, void* d_ws, size_t ws_size,
                              hipStream_t stream) {
    (void)in_sizes; (void)n_in; (void)d_ws; (void)ws_size; (void)out_size;
    const float* inputs = (const float*)d_in[0];
    const float* W_ih_e = (const float*)d_in[1];
    const float* W_hh_e = (const float*)d_in[2];
    const float* b_e    = (const float*)d_in[3];
    const float* W_ih_d = (const float*)d_in[4];
    const float* W_hh_d = (const float*)d_in[5];
    const float* b_d    = (const float*)d_in[6];
    const float* W_fc   = (const float*)d_in[7];
    const float* b_fc   = (const float*)d_in[8];
    float* out = (float*)d_out;

    dim3 grid(BATCH / MTILE);   // 64 independent batch tiles, no inter-WG sync
    dim3 block(128);            // 4 wave32 waves per workgroup
    lstm_seq2seq_kernel<<<grid, block, 0, stream>>>(
        inputs, W_ih_e, W_hh_e, b_e, W_ih_d, W_hh_d, b_d, W_fc, b_fc, out);
}
